// PA_MSA_51496657879515
// MI455X (gfx1250) — matrix operand order repaired and checked by reading, not compiled, not run on hardware
//
#include <hip/hip_runtime.h>
#include <hip/hip_bf16.h>

typedef __attribute__((ext_vector_type(16))) __bf16 v16bf;
typedef __attribute__((ext_vector_type(8)))  __bf16 v8bf;
typedef __attribute__((ext_vector_type(8)))  float  v8f;
typedef __attribute__((ext_vector_type(4)))  float  f4;
typedef __attribute__((ext_vector_type(4)))  int    v4i;

#define WMMA_BF16(a, b, c) \
  __builtin_amdgcn_wmma_f32_16x16x32_bf16(false, (a), false, (b), (short)0, (c), false, false)

// ---------------------------------------------------------------------------
// CDNA5 async global->LDS copy (ASYNCcnt-tracked). The builtin takes exact
// address-space-qualified int4 pointers: (AS1 v4i* gsrc, AS3 v4i* ldsdst,
// imm offset, imm cpol). C-style casts perform the addrspacecast.
// ---------------------------------------------------------------------------
#if defined(__AMDGCN__) && \
    __has_builtin(__builtin_amdgcn_global_load_async_to_lds_b128) && \
    __has_builtin(__builtin_amdgcn_s_wait_asynccnt)
#define USE_ASYNC_LDS 1
#define ASYNC_CP16(ldsp, gp)                                                   \
  __builtin_amdgcn_global_load_async_to_lds_b128(                              \
      (__attribute__((address_space(1))) v4i*)(gp),                            \
      (__attribute__((address_space(3))) v4i*)(ldsp), 0, 0)
#define ASYNC_WAIT() __builtin_amdgcn_s_wait_asynccnt(0)
#else
#define USE_ASYNC_LDS 0
#define ASYNC_CP16(ldsp, gp) (*(v8bf*)(ldsp) = *(const v8bf*)(gp))
#define ASYNC_WAIT() ((void)0)
#endif

// LDS row strides padded to 16B multiples (in bf16 elements)
#define AS_STRIDE 40   // 32 + 8
#define BS_STRIDE 72   // 64 + 8
#define QS_STRIDE 40
#define AV_STRIDE 72
#define VS_STRIDE 136  // 128 + 8

// ---------------------------------------------------------------------------
// GEMM: C[b] = A (MxK, f32 weights) * B[b] (KxN activations), bf16 output.
// Block tile 64x64, 256 threads = 8 waves, each wave -> 32x16 (2 accums).
// Every thread stages exactly one 16B chunk of A and one of B per K-step.
// ---------------------------------------------------------------------------
template <bool B_F32, bool CAT>
__global__ __launch_bounds__(256) void gemm_pw_kernel(
    const float* __restrict__ A, const void* __restrict__ B0,
    const void* __restrict__ B1, __bf16* __restrict__ C,
    int M, int N, int K)
{
  __shared__ __bf16 As[64 * AS_STRIDE];   // 64 x 32 (padded)
  __shared__ __bf16 Bs[32 * BS_STRIDE];   // 32 x 64 (padded)

  const int tid  = threadIdx.x;
  const int lane = tid & 31, wave = tid >> 5;
  const int wm = wave >> 2, wn = wave & 3;       // 2 x 4 wave grid
  const int l  = lane & 15, hi = lane >> 4;

  const int m_base = blockIdx.y * 64;
  const int n_base = blockIdx.x * 64;
  const int batch  = blockIdx.z;
  const int Ksrc   = CAT ? (K >> 1) : K;
  const long bOff  = (long)batch * Ksrc * N;

  // A chunk: 64x32 = 256 chunks of 8 -> one per thread
  const int ar = tid >> 2, ac = (tid & 3) * 8;
  // B chunk: 32x64 = 256 chunks of 8 -> one per thread
  const int br = tid >> 3, bc = (tid & 7) * 8;

  v8f acc0 = {}, acc1 = {};

  for (int k0 = 0; k0 < K; k0 += 32) {
    // ---- stage A tile (f32 weights -> bf16), 2x b128 load + 1x b128 store
    {
      const float* ap = A + (long)(m_base + ar) * K + k0 + ac;
      f4 lo = *(const f4*)ap;
      f4 hc = *(const f4*)(ap + 4);
      v8bf v;
      #pragma unroll
      for (int i = 0; i < 4; ++i) { v[i] = (__bf16)lo[i]; v[4 + i] = (__bf16)hc[i]; }
      *(v8bf*)&As[ar * AS_STRIDE + ac] = v;
    }
    // ---- stage B tile
    {
      int k = k0 + br;
      if (CAT) {
        const __bf16* src = (k < Ksrc) ? (const __bf16*)B0 : (const __bf16*)B1;
        int kk = (k < Ksrc) ? k : (k - Ksrc);
        ASYNC_CP16(&Bs[br * BS_STRIDE + bc], src + bOff + (long)kk * N + n_base + bc);
      } else if (B_F32) {
        const float* bp = (const float*)B0 + bOff + (long)k * N + n_base + bc;
        f4 lo = *(const f4*)bp;
        f4 hc = *(const f4*)(bp + 4);
        v8bf v;
        #pragma unroll
        for (int i = 0; i < 4; ++i) { v[i] = (__bf16)lo[i]; v[4 + i] = (__bf16)hc[i]; }
        *(v8bf*)&Bs[br * BS_STRIDE + bc] = v;
      } else {
        ASYNC_CP16(&Bs[br * BS_STRIDE + bc],
                   (const __bf16*)B0 + bOff + (long)k * N + n_base + bc);
      }
    }
    ASYNC_WAIT();
    __syncthreads();

    // fragments per CDNA5 16-bit A layout (lanes<16: K{0-7,16-23}; lanes>=16: K{8-15,24-31})
    v16bf a0, a1, bfrag;
    const int m0 = wm * 32;
    #pragma unroll
    for (int i = 0; i < 8; ++i) {
      a0[i]     = As[(m0 + l) * AS_STRIDE + hi * 8 + i];
      a0[8 + i] = As[(m0 + l) * AS_STRIDE + 16 + hi * 8 + i];
      a1[i]     = As[(m0 + 16 + l) * AS_STRIDE + hi * 8 + i];
      a1[8 + i] = As[(m0 + 16 + l) * AS_STRIDE + 16 + hi * 8 + i];
    }
    #pragma unroll
    for (int i = 0; i < 16; ++i)  // B: lanes<16 hold K{0-7,16-23}, lanes>=16 hold K{8-15,24-31}
      bfrag[i] = Bs[((i >> 3) * 16 + hi * 8 + (i & 7)) * BS_STRIDE + wn * 16 + l];

    acc0 = WMMA_BF16(a0, bfrag, acc0);
    acc1 = WMMA_BF16(a1, bfrag, acc1);
    __syncthreads();
  }

  const long cOff = (long)batch * M * N;
  const int  n    = n_base + wn * 16 + l;
  #pragma unroll
  for (int r = 0; r < 8; ++r) {
    int m = m_base + wm * 32 + (hi ? 8 + r : r);
    C[cOff + (long)m * N + n]        = (__bf16)acc0[r];
    C[cOff + (long)(m + 16) * N + n] = (__bf16)acc1[r];
  }
}

// ---------------------------------------------------------------------------
// Depthwise 3x3, SAME padding. 8 pixels per thread: one b128 center load and
// two scalar halo loads per row, one b128 store.
// ---------------------------------------------------------------------------
__global__ __launch_bounds__(256) void dw3x3_kernel(
    const __bf16* __restrict__ in, const float* __restrict__ w,
    __bf16* __restrict__ out, int C, int H, int W)
{
  long chunk = (long)blockIdx.x * 256 + threadIdx.x;   // 8 pixels per chunk
  const int cw = W >> 3;
  int x0 = (int)(chunk % cw) * 8; long r = chunk / cw;
  int y = (int)(r % H); r /= H;
  int c = (int)(r % C); int b = (int)(r / C);

  const float*  wc = w + c * 9;
  const __bf16* p  = in + ((long)(b * C + c)) * H * W;

  float row[3][10];
  #pragma unroll
  for (int dy = 0; dy < 3; ++dy) {
    int yy = y + dy - 1;
    if (yy < 0 || yy >= H) {
      #pragma unroll
      for (int i = 0; i < 10; ++i) row[dy][i] = 0.f;
    } else {
      const __bf16* pr = p + (long)yy * W + x0;
      row[dy][0] = (x0 > 0) ? (float)pr[-1] : 0.f;
      v8bf cen = *(const v8bf*)pr;
      #pragma unroll
      for (int i = 0; i < 8; ++i) row[dy][1 + i] = (float)cen[i];
      row[dy][9] = (x0 + 8 < W) ? (float)pr[8] : 0.f;
    }
  }

  v8bf o;
  #pragma unroll
  for (int i = 0; i < 8; ++i) {
    float acc = 0.f;
    #pragma unroll
    for (int dy = 0; dy < 3; ++dy)
      #pragma unroll
      for (int dx = 0; dx < 3; ++dx)
        acc += row[dy][i + dx] * wc[dy * 3 + dx];
    o[i] = (__bf16)acc;
  }
  *(v8bf*)(out + chunk * 8) = o;
}

// ---------------------------------------------------------------------------
// Per-row (length N) L2 norm -> reciprocal (folded into gram scaling).
// ---------------------------------------------------------------------------
__global__ __launch_bounds__(256) void l2norm_kernel(
    const __bf16* __restrict__ in, float* __restrict__ inv, int N)
{
  const int row = blockIdx.x;
  const __bf16* p = in + (long)row * N;
  float s = 0.f;
  for (int i = threadIdx.x * 8; i < N; i += 256 * 8) {
    v8bf v = *(const v8bf*)(p + i);
    #pragma unroll
    for (int j = 0; j < 8; ++j) { float f = (float)v[j]; s += f * f; }
  }
  __shared__ float red[256];
  red[threadIdx.x] = s; __syncthreads();
  for (int st = 128; st > 0; st >>= 1) {
    if (threadIdx.x < st) red[threadIdx.x] += red[threadIdx.x + st];
    __syncthreads();
  }
  if (threadIdx.x == 0) inv[row] = 1.0f / fmaxf(sqrtf(red[0]), 1e-12f);
}

// ---------------------------------------------------------------------------
// Gram matrix S = (q/||q||)(k/||k||)^T * temp, fused softmax.
// One block per (b,h): 288 threads = 9 waves, each wave one 16x16 tile of 48x48.
// Tiles staged via ASYNCcnt-tracked global->LDS copies (one 16B chunk/thread).
// ---------------------------------------------------------------------------
__global__ __launch_bounds__(288) void gram_softmax_kernel(
    const __bf16* __restrict__ q, const __bf16* __restrict__ k,
    const float* __restrict__ invq, const float* __restrict__ invk,
    const float* __restrict__ temp, __bf16* __restrict__ attn, int N)
{
  const int bh = blockIdx.x;
  const int b = bh >> 3, h = bh & 7;

  __shared__ __bf16 Qs[48 * QS_STRIDE];
  __shared__ __bf16 Ks[48 * QS_STRIDE];
  __shared__ float  Sf[48][48];

  const int tid = threadIdx.x, lane = tid & 31, wave = tid >> 5;  // 9 waves
  const int mi = wave / 3, ni = wave % 3;
  const int l = lane & 15, hi = lane >> 4;

  const __bf16* qb = q + ((long)(b * 384 + h * 48)) * N;
  const __bf16* kb = k + ((long)(b * 384 + h * 48)) * N;

  // 48x32 tile = 192 chunks of 8; threads 0..191 -> Q, threads 96..287 -> K
  const int qi = tid, qr = qi >> 2, qc = (qi & 3) * 8;
  const int ki = tid - 96, kr = ki >> 2, kc = (ki & 3) * 8;

  v8f acc = {};
  for (int k0 = 0; k0 < N; k0 += 32) {
    if (tid < 192)  ASYNC_CP16(&Qs[qr * QS_STRIDE + qc], qb + (long)qr * N + k0 + qc);
    if (tid >= 96)  ASYNC_CP16(&Ks[kr * QS_STRIDE + kc], kb + (long)kr * N + k0 + kc);
    ASYNC_WAIT();
    __syncthreads();

    v16bf a, bb;
    #pragma unroll
    for (int i = 0; i < 8; ++i) {
      a[i]      = Qs[(mi * 16 + l) * QS_STRIDE + hi * 8 + i];
      a[8 + i]  = Qs[(mi * 16 + l) * QS_STRIDE + 16 + hi * 8 + i];
      // B = K^T : B[kk, n] = K[n, kk]  (row-major read of Ks)
      bb[i]     = Ks[(ni * 16 + l) * QS_STRIDE + hi * 8 + i];
      bb[8 + i] = Ks[(ni * 16 + l) * QS_STRIDE + 16 + hi * 8 + i];
    }
    acc = WMMA_BF16(a, bb, acc);
    __syncthreads();
  }

  const float tmprt = temp[h];
  const int n = ni * 16 + l;
  #pragma unroll
  for (int r = 0; r < 8; ++r) {
    int m = mi * 16 + (hi ? 8 + r : r);
    Sf[m][n] = acc[r] * invq[b * 384 + h * 48 + m] * invk[b * 384 + h * 48 + n] * tmprt;
  }
  __syncthreads();

  if (tid < 48) {
    float mx = -1e30f;
    for (int j = 0; j < 48; ++j) mx = fmaxf(mx, Sf[tid][j]);
    float s = 0.f;
    for (int j = 0; j < 48; ++j) s += __expf(Sf[tid][j] - mx);
    float is = 1.0f / s;
    __bf16* arow = attn + ((long)bh * 48 + tid) * 64;
    for (int j = 0; j < 48; ++j) arow[j] = (__bf16)(__expf(Sf[tid][j] - mx) * is);
    for (int j = 48; j < 64; ++j) arow[j] = (__bf16)0.0f;
  }
}

// ---------------------------------------------------------------------------
// out[b,h] = attn(48x48, padded to 48x64) @ v(48xN), f32 output.
// Block = 48x128 output tile, 8 waves (one N column of 16 each, 3 M accums).
// ---------------------------------------------------------------------------
__global__ __launch_bounds__(256) void attn_out_kernel(
    const __bf16* __restrict__ attn, const __bf16* __restrict__ v,
    float* __restrict__ out, int N)
{
  const int bh = blockIdx.y;
  const int b = bh >> 3, h = bh & 7;
  const int n_base = blockIdx.x * 128;

  __shared__ __bf16 Avs[48 * AV_STRIDE];   // 48 x 64 (padded)
  __shared__ __bf16 Vs[32 * VS_STRIDE];    // 32 x 128 (padded)

  const int tid = threadIdx.x, lane = tid & 31, wave = tid >> 5;
  const int l = lane & 15, hi = lane >> 4;

  // attn tile: 48x64 = 384 chunks of 8; thread does chunk tid (+ tid+256 if tid<128)
  const __bf16* arow = attn + (long)bh * 48 * 64;
  {
    int e = tid, r = e >> 3, c = (e & 7) * 8;
    ASYNC_CP16(&Avs[r * AV_STRIDE + c], arow + r * 64 + c);
    if (tid < 128) {
      e = tid + 256; r = e >> 3; c = (e & 7) * 8;
      ASYNC_CP16(&Avs[r * AV_STRIDE + c], arow + r * 64 + c);
    }
  }

  const __bf16* vb = v + ((long)(b * 384 + h * 48)) * N;
  v8f acc0 = {}, acc1 = {}, acc2 = {};
  v8bf zero = {};

  for (int k0 = 0; k0 < 64; k0 += 32) {
    // V tile: 32x128 = 512 chunks of 8 -> 2 per thread; rows >=48 are zero pad
    #pragma unroll
    for (int half = 0; half < 2; ++half) {
      int e = tid + half * 256;
      int r = e >> 4, c = (e & 15) * 8;
      int kk = k0 + r;
      if (kk < 48) ASYNC_CP16(&Vs[r * VS_STRIDE + c], vb + (long)kk * N + n_base + c);
      else         *(v8bf*)&Vs[r * VS_STRIDE + c] = zero;
    }
    ASYNC_WAIT();
    __syncthreads();

    v16bf a0, a1, a2, bb;
    #pragma unroll
    for (int i = 0; i < 8; ++i) {
      a0[i]     = Avs[l * AV_STRIDE + k0 + hi * 8 + i];
      a0[8 + i] = Avs[l * AV_STRIDE + k0 + 16 + hi * 8 + i];
      a1[i]     = Avs[(16 + l) * AV_STRIDE + k0 + hi * 8 + i];
      a1[8 + i] = Avs[(16 + l) * AV_STRIDE + k0 + 16 + hi * 8 + i];
      a2[i]     = Avs[(32 + l) * AV_STRIDE + k0 + hi * 8 + i];
      a2[8 + i] = Avs[(32 + l) * AV_STRIDE + k0 + 16 + hi * 8 + i];
    }
    #pragma unroll
    for (int i = 0; i < 16; ++i)
      bb[i] = Vs[((i >> 3) * 16 + hi * 8 + (i & 7)) * VS_STRIDE + wave * 16 + l];

    acc0 = WMMA_BF16(a0, bb, acc0);
    acc1 = WMMA_BF16(a1, bb, acc1);
    acc2 = WMMA_BF16(a2, bb, acc2);
    __syncthreads();
  }

  float* ob = out + ((long)(b * 384 + h * 48)) * N;
  const int n = n_base + wave * 16 + l;
  #pragma unroll
  for (int r = 0; r < 8; ++r) {
    int m = hi ? 8 + r : r;
    ob[(long)m * N + n]        = acc0[r];
    ob[(long)(16 + m) * N + n] = acc1[r];
    ob[(long)(32 + m) * N + n] = acc2[r];
  }
}

// ---------------------------------------------------------------------------
extern "C" void kernel_launch(void* const* d_in, const int* in_sizes, int n_in,
                              void* d_out, int out_size, void* d_ws, size_t ws_size,
                              hipStream_t stream) {
  const float* x       = (const float*)d_in[0];
  const float* t       = (const float*)d_in[1];
  const float* w_q     = (const float*)d_in[2];
  const float* w_q_dw  = (const float*)d_in[3];
  const float* w_qT    = (const float*)d_in[4];
  const float* w_qT_dw = (const float*)d_in[5];
  const float* w_qcat  = (const float*)d_in[6];
  const float* w_k     = (const float*)d_in[7];
  const float* w_k_dw  = (const float*)d_in[8];
  const float* w_v     = (const float*)d_in[9];
  const float* w_v_dw  = (const float*)d_in[10];
  const float* temp    = (const float*)d_in[11];
  float* out = (float*)d_out;

  const int Bn = 4, C = 384, H = 128, W = 128, HEADS = 8;
  const int N = H * W;
  const size_t elems = (size_t)Bn * C * N;

  // workspace layout: 4 bf16 activation buffers + norms + padded attn
  __bf16* buf0 = (__bf16*)d_ws;
  __bf16* buf1 = buf0 + elems;
  __bf16* buf2 = buf1 + elems;
  __bf16* buf3 = buf2 + elems;
  float*  invq = (float*)(buf3 + elems);
  float*  invk = invq + Bn * C;
  __bf16* attn = (__bf16*)(invk + Bn * C);   // (Bn*HEADS, 48, 64)

  dim3 gGemm(N / 64, C / 64, Bn);
  dim3 b256(256);
  const unsigned dwBlocks = (unsigned)(elems / 8 / 256);

  // q path: pw(x,w_q) -> dw -> buf1
  gemm_pw_kernel<true,  false><<<gGemm, b256, 0, stream>>>(w_q,  x,   nullptr, buf0, C, N, C);
  dw3x3_kernel<<<dwBlocks, b256, 0, stream>>>(buf0, w_q_dw, buf1, C, H, W);
  // qT path: pw(t,w_qT) -> dw -> buf2
  gemm_pw_kernel<true,  false><<<gGemm, b256, 0, stream>>>(w_qT, t,   nullptr, buf0, C, N, C);
  dw3x3_kernel<<<dwBlocks, b256, 0, stream>>>(buf0, w_qT_dw, buf2, C, H, W);
  // qcat fusion: concat(buf1,buf2) x w_qcat -> buf0 (q for attention)
  gemm_pw_kernel<false, true ><<<gGemm, b256, 0, stream>>>(w_qcat, buf1, buf2, buf0, C, N, 2 * C);
  // k path -> buf3
  gemm_pw_kernel<true,  false><<<gGemm, b256, 0, stream>>>(w_k,  x,   nullptr, buf1, C, N, C);
  dw3x3_kernel<<<dwBlocks, b256, 0, stream>>>(buf1, w_k_dw, buf3, C, H, W);
  // v path -> buf2 (t1d no longer needed)
  gemm_pw_kernel<true,  false><<<gGemm, b256, 0, stream>>>(w_v,  x,   nullptr, buf1, C, N, C);
  dw3x3_kernel<<<dwBlocks, b256, 0, stream>>>(buf1, w_v_dw, buf2, C, H, W);

  // reciprocal L2 norms (folded into gram scaling)
  l2norm_kernel<<<Bn * C, b256, 0, stream>>>(buf0, invq, N);
  l2norm_kernel<<<Bn * C, b256, 0, stream>>>(buf3, invk, N);

  // 48x48 gram + temperature + softmax per (b,h)
  gram_softmax_kernel<<<Bn * HEADS, 288, 0, stream>>>(buf0, buf3, invq, invk, temp, attn, N);

  // out = attn @ v, f32 straight to d_out
  attn_out_kernel<<<dim3(N / 128, Bn * HEADS), b256, 0, stream>>>(attn, buf2, out, N);
}
